// TTT_43774306680900
// MI455X (gfx1250) — matrix operand order repaired and checked by reading, not compiled, not run on hardware
//
#include <hip/hip_runtime.h>
#include <hip/hip_bf16.h>

typedef __attribute__((ext_vector_type(16))) __bf16 v16bf;
typedef __attribute__((ext_vector_type(8)))  float  v8f;

#define DIM   768
#define HEADS 12
#define HD    64
#define BATCH 16
#define NTOK  1024       // 32*32
#define NQKV  2496       // 3*768 + 3*64
#define NCAT  832        // 768 + 64
#define LRATE 0.001f

__device__ __forceinline__ __bf16 f2bf(float f) {
    unsigned u = __builtin_bit_cast(unsigned, f);
    u += 0x7FFFu + ((u >> 16) & 1u);            // round-to-nearest-even
    unsigned short h = (unsigned short)(u >> 16);
    return __builtin_bit_cast(__bf16, h);
}

__device__ __forceinline__ v8f wmma_bf16(v16bf a, v16bf b, v8f c) {
    return __builtin_amdgcn_wmma_f32_16x16x32_bf16(
        false, a, false, b, (short)0, c, false, false);
}

// ---- fragment loaders -------------------------------------------------------
// A (16x32 bf16, row-major src): lane<16 holds K in {0..15}, lane>=16 in {16..31}
__device__ __forceinline__ v16bf load_fragA(const __bf16* __restrict__ A, int lda,
                                            int row, int k0, int hi) {
    union { v16bf v; uint4 u[2]; } r;
    const __bf16* p = A + (size_t)row * lda + k0 + hi * 16;
    r.u[0] = *(const uint4*)p;
    r.u[1] = *(const uint4*)(p + 8);
    return r.v;
}
// A fragment, converting from f32 source on the fly
__device__ __forceinline__ v16bf load_fragA_f32(const float* __restrict__ A, int lda,
                                                int row, int k0, int hi) {
    union { v16bf v; __bf16 h[16]; } r;
    const float* p = A + (size_t)row * lda + k0 + hi * 16;
#pragma unroll
    for (int i = 0; i < 8; i++) { r.h[i] = f2bf(p[i]); r.h[8 + i] = f2bf(p[8 + i]); }
    return r.v;
}
// B fragment from pre-packed buffer: frag (nt*KT+kt) = 512 contiguous halves, 16 per lane
__device__ __forceinline__ v16bf load_fragP(const __bf16* __restrict__ Bp,
                                            int nt, int kt, int KT, int lane) {
    union { v16bf v; uint4 u[2]; } r;
    const __bf16* p = Bp + (((size_t)nt * KT + kt) << 9) + lane * 16;
    r.u[0] = *(const uint4*)p;
    r.u[1] = *(const uint4*)(p + 8);
    return r.v;
}
// B fragment from B^T stored row-major (col-contiguous-in-K):
// B[k][col] = T[col*ldt + k]; lanes 0..15 K=k0..k0+15, lanes 16..31 K=k0+16..k0+31
__device__ __forceinline__ v16bf load_fragBT(const __bf16* __restrict__ T, int ldt,
                                             int col, int k0, int hi) {
    union { v16bf v; uint4 u[2]; } r;
    const __bf16* p = T + (size_t)col * ldt + k0 + hi * 16;
    r.u[0] = *(const uint4*)p;
    r.u[1] = *(const uint4*)(p + 8);
    return r.v;
}

// ---- utility kernels --------------------------------------------------------
__global__ void cast_bf16(const float* __restrict__ in, __bf16* __restrict__ out, int n) {
    int i = blockIdx.x * blockDim.x + threadIdx.x;
    if (i < n) out[i] = f2bf(in[i]);
}

__global__ void zero_f32(float* p, int n) {
    int i = blockIdx.x * blockDim.x + threadIdx.x;
    if (i < n) p[i] = 0.0f;
}

// Pack row-major W[K,N] (f32) into B-fragment-major bf16
__global__ void pack_weight(const float* __restrict__ W, __bf16* __restrict__ out,
                            int K, int N) {
    int o = blockIdx.x * blockDim.x + threadIdx.x;
    if (o >= K * N) return;
    int s = o & 15, lane = (o >> 4) & 31, frag = o >> 9;
    int KT = K >> 5;
    int kt = frag % KT, nt = frag / KT;
    int n = lane & 15, hi = lane >> 4;
    int k = kt * 32 + hi * 16 + s;
    out[o] = f2bf(W[(size_t)k * N + nt * 16 + n]);
}

// Pack per-head weight [12,64,64] -> fragments; optional SGD-with-norm update fused in.
__global__ void pack_head_weight(const float* __restrict__ W, const float* __restrict__ G,
                                 const float* __restrict__ ss, __bf16* __restrict__ out) {
    int o = blockIdx.x * blockDim.x + threadIdx.x;
    if (o >= HEADS * 8 * 512) return;
    int s = o & 15, lane = (o >> 4) & 31, frag = o >> 9;  // frag = (H*4+nt)*2+kt
    int kt = frag & 1, nt = (frag >> 1) & 3, H = frag >> 3;
    int n = lane & 15, hi = lane >> 4;
    int k = kt * 32 + hi * 16 + s;
    int idx = H * 4096 + k * 64 + nt * 16 + n;
    float v = W[idx];
    if (G) v -= LRATE * G[idx] / (sqrtf(*ss) + 1.0f);
    out[o] = f2bf(v);
}

// ---- big GEMM: C[M,N] f32 = A[M,K] bf16 @ Bpacked + bias; 64x64 per wave ----
__global__ void __launch_bounds__(256)
gemm_bf16_f32out(const __bf16* __restrict__ A, const __bf16* __restrict__ Bp,
                 const float* __restrict__ bias, float* __restrict__ C,
                 int M, int N, int K) {
    int wid  = (blockIdx.x * blockDim.x + threadIdx.x) >> 5;
    int lane = threadIdx.x & 31, l16 = lane & 15, hi = lane >> 4;
    int nBlk = N >> 6;
    int mb = wid / nBlk, nb = wid % nBlk;
    if (mb >= (M >> 6)) return;
    int m0 = mb << 6, n0 = nb << 6, KT = K >> 5;
    v8f acc[4][4];
#pragma unroll
    for (int i = 0; i < 4; i++)
#pragma unroll
        for (int j = 0; j < 4; j++) acc[i][j] = (v8f)(0.0f);
    for (int kt = 0; kt < KT; ++kt) {
        int k0 = kt << 5;
        v16bf af[4], bfr[4];
#pragma unroll
        for (int i = 0; i < 4; i++) af[i] = load_fragA(A, K, m0 + i * 16 + l16, k0, hi);
#pragma unroll
        for (int j = 0; j < 4; j++) bfr[j] = load_fragP(Bp, (n0 >> 4) + j, kt, KT, lane);
#pragma unroll
        for (int i = 0; i < 4; i++)
#pragma unroll
            for (int j = 0; j < 4; j++) acc[i][j] = wmma_bf16(af[i], bfr[j], acc[i][j]);
    }
    float bv[4];
#pragma unroll
    for (int j = 0; j < 4; j++) bv[j] = bias[n0 + j * 16 + l16];
#pragma unroll
    for (int i = 0; i < 4; i++)
#pragma unroll
        for (int j = 0; j < 4; j++)
#pragma unroll
            for (int r = 0; r < 8; r++) {
                int row = m0 + i * 16 + hi * 8 + r;
                C[(size_t)row * N + n0 + j * 16 + l16] = acc[i][j][r] + bv[j];
            }
}

// ---- transpose k1 slice of qkv into bf16 [b,H,d,n] (= [b*768+d, 1024]) ------
__global__ void __launch_bounds__(256)
transpose_k1(const float* __restrict__ qkv, __bf16* __restrict__ k1T) {
    __shared__ float lds[16][17];
    int blk = blockIdx.x;
    int dt = blk % 48, nt = (blk / 48) & 63, b = blk / (48 * 64);
    int t = threadIdx.x, r = t >> 4, c = t & 15;
    int n = nt * 16 + r, d = dt * 16 + c;
    lds[r][c] = qkv[((size_t)(b * NTOK + n)) * NQKV + DIM + d];
    __syncthreads();
    int d2 = dt * 16 + r, n2 = nt * 16 + c;
    k1T[((size_t)(b * DIM + d2)) * NTOK + n2] = f2bf(lds[c][r]);
}

// ---- extract q2,k2,v2 planar f32 [b,64,1024] --------------------------------
__global__ void extract_spatial(const float* __restrict__ qkv, float* __restrict__ q2p,
                                float* __restrict__ k2p, float* __restrict__ v2p) {
    int idx = blockIdx.x * blockDim.x + threadIdx.x;  // 16*1024*192
    if (idx >= BATCH * NTOK * 192) return;
    int t = idx % 192, bn = idx / 192;
    int b = bn >> 10, n = bn & 1023;
    float v = qkv[(size_t)bn * NQKV + 3 * DIM + t];
    int arr = t >> 6, j = t & 63;
    float* out = (arr == 0) ? q2p : ((arr == 1) ? k2p : v2p);
    out[(size_t)(b * HD + j) * NTOK + n] = v;
}

// ---- fused TTT loss fwd + grad-of-output: writes gu^T, gs^T (bf16) ----------
__global__ void __launch_bounds__(256)
ttt_loss_fused(const float* __restrict__ qkv, const __bf16* __restrict__ a1P,
               const __bf16* __restrict__ a2P, __bf16* __restrict__ guT,
               __bf16* __restrict__ gsT) {
    int wid  = (blockIdx.x * blockDim.x + threadIdx.x) >> 5;
    int lane = threadIdx.x & 31, l16 = lane & 15, hi = lane >> 4;
    int ntl = wid & 63, H = (wid >> 6) % HEADS, b = wid / (64 * HEADS);
    if (b >= BATCH) return;
    int n0 = ntl * 16;
    const float* Ab = qkv + (size_t)(b * NTOK) * NQKV;
    int cA = DIM + H * HD;  // k1 columns
    v8f uacc[4], sacc[4];
#pragma unroll
    for (int nt = 0; nt < 4; nt++) { uacc[nt] = (v8f)(0.0f); sacc[nt] = (v8f)(0.0f); }
#pragma unroll
    for (int kt = 0; kt < 2; kt++) {
        v16bf af = load_fragA_f32(Ab, NQKV, n0 + l16, cA + kt * 32, hi);
#pragma unroll
        for (int nt = 0; nt < 4; nt++) {
            v16bf b1 = load_fragP(a1P, H * 4 + nt, kt, 2, lane);
            v16bf b2 = load_fragP(a2P, H * 4 + nt, kt, 2, lane);
            uacc[nt] = wmma_bf16(af, b1, uacc[nt]);
            sacc[nt] = wmma_bf16(af, b2, sacc[nt]);
        }
    }
    const float invM1 = 1.0f / 12582912.0f;  // B*H*n*d
#pragma unroll
    for (int nt = 0; nt < 4; nt++) {
        int dcol = H * HD + nt * 16 + l16;
        union { uint4 q; __bf16 h[8]; } pu, ps;
#pragma unroll
        for (int i = 0; i < 8; i++) {
            int nrow = n0 + hi * 8 + i;
            float u = uacc[nt][i], s = sacc[nt][i];
            float sig = 1.0f / (1.0f + __expf(-s));
            float g = s * sig;
            float v = qkv[((size_t)(b * NTOK + nrow)) * NQKV + 2 * DIM + dcol];
            float r = (u * g - v) * invM1;
            pu.h[i] = f2bf(r * g);
            ps.h[i] = f2bf(r * u * sig * (1.0f + s * (1.0f - sig)));
        }
        size_t base = ((size_t)((b * HEADS + H) * HD + dcol)) * NTOK + n0 + hi * 8;
        *(uint4*)(guT + base) = pu.q;
        *(uint4*)(gsT + base) = ps.q;
    }
}

// ---- grad GEMMs: grad1/2[H] += k1^T @ gu/gs, split-K over batch -------------
__global__ void __launch_bounds__(256)
grad_gemm(const __bf16* __restrict__ k1T, const __bf16* __restrict__ guT,
          const __bf16* __restrict__ gsT, float* __restrict__ grad1,
          float* __restrict__ grad2) {
    int wid  = (blockIdx.x * blockDim.x + threadIdx.x) >> 5;
    int lane = threadIdx.x & 31, l16 = lane & 15, hi = lane >> 4;
    int b = wid & 15, nt = (wid >> 4) & 3, mt = (wid >> 6) & 3, H = wid >> 8;
    if (H >= HEADS) return;
    size_t hb = (size_t)(b * HEADS + H) * HD;
    const __bf16* Ab = k1T + hb * NTOK;
    const __bf16* Gu = guT + hb * NTOK;
    const __bf16* Gs = gsT + hb * NTOK;
    v8f a1 = (v8f)(0.0f), a2 = (v8f)(0.0f);
    for (int kt = 0; kt < 32; kt++) {
        int k0 = kt * 32;
        v16bf af = load_fragA(Ab, NTOK, mt * 16 + l16, k0, hi);
        v16bf b1 = load_fragBT(Gu, NTOK, nt * 16 + l16, k0, hi);
        v16bf b2 = load_fragBT(Gs, NTOK, nt * 16 + l16, k0, hi);
        a1 = wmma_bf16(af, b1, a1);
        a2 = wmma_bf16(af, b2, a2);
    }
#pragma unroll
    for (int i = 0; i < 8; i++) {
        int m = mt * 16 + hi * 8 + i, j = nt * 16 + l16;
        atomicAdd(grad1 + H * 4096 + m * 64 + j, a1[i]);
        atomicAdd(grad2 + H * 4096 + m * 64 + j, a2[i]);
    }
}

// ---- depthwise conv residual + grad of a3 -----------------------------------
__global__ void __launch_bounds__(256)
conv_grad(const float* __restrict__ k2p, const float* __restrict__ v2p,
          const float* __restrict__ w3, float* __restrict__ grad3) {
    int bc = blockIdx.x, c = bc % HD, b = bc / HD;
    const float* K2 = k2p + (size_t)(b * HD + c) * NTOK;
    const float* V2 = v2p + (size_t)(b * HD + c) * NTOK;
    float wl[9], part[9];
#pragma unroll
    for (int i = 0; i < 9; i++) { wl[i] = w3[c * 9 + i]; part[i] = 0.0f; }
    const float invM2 = 1.0f / 1048576.0f;  // B*d*h*w
    for (int p = threadIdx.x; p < NTOK; p += 256) {
        int y = p >> 5, x = p & 31;
        float f = 0.0f, nb[9];
#pragma unroll
        for (int dy = 0; dy < 3; dy++)
#pragma unroll
            for (int dx = 0; dx < 3; dx++) {
                int yy = y + dy - 1, xx = x + dx - 1;
                float kv = (yy >= 0 && yy < 32 && xx >= 0 && xx < 32) ? K2[yy * 32 + xx] : 0.0f;
                nb[dy * 3 + dx] = kv;
                f += wl[dy * 3 + dx] * kv;
            }
        float r = (f - V2[p]) * invM2;
#pragma unroll
        for (int i = 0; i < 9; i++) part[i] += r * nb[i];
    }
    __shared__ float red[256];
    for (int i = 0; i < 9; i++) {
        red[threadIdx.x] = part[i];
        __syncthreads();
        for (int s = 128; s > 0; s >>= 1) {
            if ((int)threadIdx.x < s) red[threadIdx.x] += red[threadIdx.x + s];
            __syncthreads();
        }
        if (threadIdx.x == 0) atomicAdd(grad3 + c * 9 + i, red[0]);
        __syncthreads();
    }
}

__global__ void sumsq(const float* __restrict__ a, int n, float* __restrict__ out) {
    __shared__ float red[256];
    float s = 0.0f;
    for (int i = blockIdx.x * 256 + threadIdx.x; i < n; i += gridDim.x * 256) {
        float v = a[i]; s += v * v;
    }
    red[threadIdx.x] = s;
    __syncthreads();
    for (int k = 128; k > 0; k >>= 1) {
        if ((int)threadIdx.x < k) red[threadIdx.x] += red[threadIdx.x + k];
        __syncthreads();
    }
    if (threadIdx.x == 0) atomicAdd(out, red[0]);
}

__global__ void upd_conv(const float* __restrict__ w3, const float* __restrict__ g3,
                         const float* __restrict__ ss, float* __restrict__ mw3) {
    int i = blockIdx.x * blockDim.x + threadIdx.x;
    if (i < HD * 9) mw3[i] = w3[i] - LRATE * g3[i] / (sqrtf(*ss) + 1.0f);
}

// ---- fused forward: x1 = (q1@mw1)*silu(q1@mw2) -> concat (bf16) -------------
__global__ void __launch_bounds__(256)
ttt_fwd_fused(const float* __restrict__ qkv, const __bf16* __restrict__ m1P,
              const __bf16* __restrict__ m2P, __bf16* __restrict__ concat) {
    int wid  = (blockIdx.x * blockDim.x + threadIdx.x) >> 5;
    int lane = threadIdx.x & 31, l16 = lane & 15, hi = lane >> 4;
    int ntl = wid & 63, H = (wid >> 6) % HEADS, b = wid / (64 * HEADS);
    if (b >= BATCH) return;
    int n0 = ntl * 16;
    const float* Ab = qkv + (size_t)(b * NTOK) * NQKV;
    int cA = H * HD;  // q1 columns
    v8f uacc[4], sacc[4];
#pragma unroll
    for (int nt = 0; nt < 4; nt++) { uacc[nt] = (v8f)(0.0f); sacc[nt] = (v8f)(0.0f); }
#pragma unroll
    for (int kt = 0; kt < 2; kt++) {
        v16bf af = load_fragA_f32(Ab, NQKV, n0 + l16, cA + kt * 32, hi);
#pragma unroll
        for (int nt = 0; nt < 4; nt++) {
            v16bf b1 = load_fragP(m1P, H * 4 + nt, kt, 2, lane);
            v16bf b2 = load_fragP(m2P, H * 4 + nt, kt, 2, lane);
            uacc[nt] = wmma_bf16(af, b1, uacc[nt]);
            sacc[nt] = wmma_bf16(af, b2, sacc[nt]);
        }
    }
#pragma unroll
    for (int nt = 0; nt < 4; nt++) {
        int col = H * HD + nt * 16 + l16;
#pragma unroll
        for (int i = 0; i < 8; i++) {
            int nrow = n0 + hi * 8 + i;
            float u = uacc[nt][i], s = sacc[nt][i];
            float sig = 1.0f / (1.0f + __expf(-s));
            concat[(size_t)(b * NTOK + nrow) * NCAT + col] = f2bf(u * s * sig);
        }
    }
}

// ---- conv forward: x2 = dwconv(q2, mw3) -> concat[...,768+c] (bf16) ---------
__global__ void __launch_bounds__(256)
conv_fwd(const float* __restrict__ q2p, const float* __restrict__ mw3,
         __bf16* __restrict__ concat) {
    int blk = blockIdx.x, y = blk & 31, b = blk >> 5;
    int t = threadIdx.x, c = t & 63, xg = t >> 6;
    const float* Q = q2p + (size_t)(b * HD + c) * NTOK;
    float wl[9];
#pragma unroll
    for (int i = 0; i < 9; i++) wl[i] = mw3[c * 9 + i];
    for (int xi = 0; xi < 8; xi++) {
        int x = xg * 8 + xi;
        float f = 0.0f;
#pragma unroll
        for (int dy = 0; dy < 3; dy++)
#pragma unroll
            for (int dx = 0; dx < 3; dx++) {
                int yy = y + dy - 1, xx = x + dx - 1;
                if (yy >= 0 && yy < 32 && xx >= 0 && xx < 32)
                    f += wl[dy * 3 + dx] * Q[yy * 32 + xx];
            }
        concat[(size_t)(b * NTOK + y * 32 + x) * NCAT + DIM + c] = f2bf(f);
    }
}

// ---- host launch ------------------------------------------------------------
extern "C" void kernel_launch(void* const* d_in, const int* in_sizes, int n_in,
                              void* d_out, int out_size, void* d_ws, size_t ws_size,
                              hipStream_t stream) {
    const float* x     = (const float*)d_in[0];
    const float* Wqkv  = (const float*)d_in[3];
    const float* bqkv  = (const float*)d_in[4];
    const float* w1    = (const float*)d_in[5];
    const float* w2    = (const float*)d_in[6];
    const float* w3    = (const float*)d_in[7];
    const float* Wproj = (const float*)d_in[8];
    const float* bproj = (const float*)d_in[9];
    float* out = (float*)d_out;

    const int M = BATCH * NTOK;  // 16384
    char* base = (char*)d_ws;
    size_t off = 0;
    auto alloc = [&](size_t bytes) { void* p = base + off; off = (off + bytes + 255) & ~(size_t)255; return p; };

    __bf16* xb     = (__bf16*)alloc((size_t)M * DIM * 2);
    __bf16* WqkvP  = (__bf16*)alloc((size_t)DIM * NQKV * 2);
    __bf16* WprojP = (__bf16*)alloc((size_t)NCAT * DIM * 2);
    float*  qkv    = (float*)alloc((size_t)M * NQKV * 4);
    __bf16* a1P    = (__bf16*)alloc((size_t)HEADS * 8 * 512 * 2);
    __bf16* a2P    = (__bf16*)alloc((size_t)HEADS * 8 * 512 * 2);
    __bf16* k1T    = (__bf16*)alloc((size_t)M * DIM * 2);
    __bf16* guT    = (__bf16*)alloc((size_t)M * DIM * 2);
    __bf16* gsT    = (__bf16*)alloc((size_t)M * DIM * 2);
    float*  q2p    = (float*)alloc((size_t)BATCH * HD * NTOK * 4);
    float*  k2p    = (float*)alloc((size_t)BATCH * HD * NTOK * 4);
    float*  v2p    = (float*)alloc((size_t)BATCH * HD * NTOK * 4);
    float*  grads  = (float*)alloc((size_t)(49152 * 2 + 576 + 3) * 4);  // grad1|grad2|grad3|ss
    float*  grad1 = grads, *grad2 = grads + 49152, *grad3 = grads + 98304, *ss = grads + 98880;
    __bf16* m1P    = (__bf16*)alloc((size_t)HEADS * 8 * 512 * 2);
    __bf16* m2P    = (__bf16*)alloc((size_t)HEADS * 8 * 512 * 2);
    float*  mw3    = (float*)alloc(576 * 4);
    __bf16* concat = (__bf16*)alloc((size_t)M * NCAT * 2);
    (void)ws_size; (void)in_sizes; (void)n_in; (void)out_size;

    // 1. casts + weight packing
    cast_bf16<<<(M * DIM) / 256, 256, 0, stream>>>(x, xb, M * DIM);
    pack_weight<<<(DIM * NQKV) / 256, 256, 0, stream>>>(Wqkv, WqkvP, DIM, NQKV);
    pack_weight<<<(NCAT * DIM) / 256, 256, 0, stream>>>(Wproj, WprojP, NCAT, DIM);
    pack_head_weight<<<192, 256, 0, stream>>>(w1, nullptr, nullptr, a1P);
    pack_head_weight<<<192, 256, 0, stream>>>(w2, nullptr, nullptr, a2P);

    // 2. qkv = x @ Wqkv + bqkv   (16384 x 2496, K=768)
    gemm_bf16_f32out<<<(256 * 39) / 8, 256, 0, stream>>>(xb, WqkvP, bqkv, qkv, M, NQKV, DIM);

    // 3. layout transforms
    transpose_k1<<<16 * 64 * 48, 256, 0, stream>>>(qkv, k1T);
    extract_spatial<<<(BATCH * NTOK * 192) / 256, 256, 0, stream>>>(qkv, q2p, k2p, v2p);
    zero_f32<<<(98883 + 255) / 256, 256, 0, stream>>>(grads, 98883);

    // 4. TTT loss forward + output-grad (WMMA) and conv grad
    ttt_loss_fused<<<(BATCH * HEADS * 64) / 8, 256, 0, stream>>>(qkv, a1P, a2P, guT, gsT);
    conv_grad<<<BATCH * HD, 256, 0, stream>>>(k2p, v2p, w3, grad3);

    // 5. weight grads via split-K WMMA GEMM with atomics
    grad_gemm<<<(HEADS * 256) / 8, 256, 0, stream>>>(k1T, guT, gsT, grad1, grad2);

    // 6. Frobenius norms + updates (re-pack updated weights for WMMA)
    sumsq<<<192, 256, 0, stream>>>(grad1, 49152, ss + 0);
    sumsq<<<192, 256, 0, stream>>>(grad2, 49152, ss + 1);
    sumsq<<<3, 256, 0, stream>>>(grad3, 576, ss + 2);
    pack_head_weight<<<192, 256, 0, stream>>>(w1, grad1, ss + 0, m1P);
    pack_head_weight<<<192, 256, 0, stream>>>(w2, grad2, ss + 1, m2P);
    upd_conv<<<3, 256, 0, stream>>>(w3, grad3, ss + 2, mw3);

    // 7. forward with updated weights -> concat buffer
    ttt_fwd_fused<<<(BATCH * HEADS * 64) / 8, 256, 0, stream>>>(qkv, m1P, m2P, concat);
    conv_fwd<<<BATCH * 32, 256, 0, stream>>>(q2p, mw3, concat);

    // 8. out = concat @ Wproj + bproj   (16384 x 768, K=832)
    gemm_bf16_f32out<<<(256 * 12) / 8, 256, 0, stream>>>(concat, WprojP, bproj, out, M, DIM, NCAT);
}
